// BaselineAttention_69389491634475
// MI455X (gfx1250) — matrix operand order repaired and checked by reading, not compiled, not run on hardware
//
#include <hip/hip_runtime.h>
#include <hip/hip_bf16.h>

// ---------------------------------------------------------------------------
// MI455X (gfx1250, wave32) implementation.
//   1) cvt X -> bf16;  transpose+cvt emb_w, Wq, Wk -> bf16 [N][K]
//   2) embed GEMM (bf16 WMMA, f32 acc, async-LDS double-buffered) + bias + PE
//   3) V projection (fp32, LDS reduce)
//   4) Q/K projections: 16-head batched bf16 WMMA GEMMs (async-LDS pipelined)
//   5) flash-style attention: 128x128 score tiles via v_wmma_f32_16x16x32_bf16,
//      async-LDS double-buffered operand streaming, online softmax (scalar V)
//   6) M16 = Wo @ dec_w fused tail, out = ctx @ M16 + dec_b (fp32)
// ---------------------------------------------------------------------------

typedef __attribute__((ext_vector_type(16))) __bf16 v16bf;
typedef __attribute__((ext_vector_type(8)))  float  v8f;
typedef __attribute__((ext_vector_type(4)))  float  f32x4;
typedef __attribute__((ext_vector_type(4)))  unsigned int u32x4;

// async-LDS builtin operand types (from hipcc diagnostic: int __vector(4) ptrs)
typedef int v4i __attribute__((__vector_size__(16)));
typedef __attribute__((address_space(1))) v4i gv4i;   // global
typedef __attribute__((address_space(3))) v4i lv4i;   // LDS

union FragU { u32x4 q[2]; v16bf v; };

#define TSTR 40   // LDS tile row stride in bf16 elements (80B: 16B-aligned, bank-spread)

__device__ __forceinline__ unsigned short f2bf(float f) {
  unsigned int u = __builtin_bit_cast(unsigned int, f);
  unsigned int r = u + 0x7FFFu + ((u >> 16) & 1u);   // round-to-nearest-even
  return (unsigned short)(r >> 16);
}
__device__ __forceinline__ float bf2f(unsigned short us) {
  return __builtin_bit_cast(float, ((unsigned int)us) << 16);
}

// CDNA5 async global->LDS copy (ASYNCcnt-tracked, no VGPR staging).
// Generic->AS casts via integer: LDS offset is the low 32 bits of the shared
// aperture address (ISA 10.2).
__device__ __forceinline__ void async_ld_b128(const void* g, const void* l) {
  __builtin_amdgcn_global_load_async_to_lds_b128(
      (gv4i*)(unsigned long long)g,
      (lv4i*)(unsigned)(unsigned long long)l,
      0, 0);
}

// A fragment: 16x32 bf16, row-major LDS tile. ISA layout: lane L holds row L&15;
// v16bf element e corresponds to K = (e>>3)*16 + (lane>>4)*8 + (e&7).
__device__ __forceinline__ v16bf load_frag_a(const unsigned short* tile, int rowBase, int lane) {
  int row  = rowBase + (lane & 15);
  int half = lane >> 4;
  const unsigned short* p = tile + row * TSTR;
  FragU u;
  u.q[0] = *(const u32x4*)(p + half * 8);         // K = half*8 .. +7
  u.q[1] = *(const u32x4*)(p + 16 + half * 8);    // K = 16+half*8 .. +7
  return u.v;
}

// B fragment from transposed tile Bt[n][k]: element e = B[K=(e>>3)*16+half*8+(e&7)][col].
__device__ __forceinline__ v16bf load_frag_b(const unsigned short* tile, int colBase, int lane) {
  int col  = colBase + (lane & 15);
  int half = lane >> 4;
  const unsigned short* p = tile + col * TSTR + half * 8;
  FragU u;
  u.q[0] = *(const u32x4*)(p);
  u.q[1] = *(const u32x4*)(p + 16);
  return u.v;
}

// --------------------------- small utility kernels -------------------------

__global__ void cvt_f32_bf16(const float* __restrict__ in, unsigned short* __restrict__ out, int n) {
  int i = blockIdx.x * 256 + threadIdx.x;
  if (i < n) out[i] = f2bf(in[i]);
}

// in: f32 [R][C] row-major  ->  out: bf16 [C][R]  (32x32 LDS tile transpose)
__global__ void transpose_cvt(const float* __restrict__ in, unsigned short* __restrict__ out,
                              int R, int C, long long inz, long long outz) {
  __shared__ float t[32][33];
  in  += (long long)blockIdx.z * inz;
  out += (long long)blockIdx.z * outz;
  int c0 = blockIdx.x * 32, r0 = blockIdx.y * 32;
  for (int j = 0; j < 32; j += 8)
    t[threadIdx.y + j][threadIdx.x] = in[(size_t)(r0 + threadIdx.y + j) * C + c0 + threadIdx.x];
  __syncthreads();
  for (int j = 0; j < 32; j += 8)
    out[(size_t)(c0 + threadIdx.y + j) * R + r0 + threadIdx.x] = f2bf(t[threadIdx.x][threadIdx.y + j]);
}

// ------------------------------ WMMA GEMM ----------------------------------
// C[M][N](bf16) = A[M][K](bf16, row-major) x Bt[N][K](bf16, i.e. B transposed)
// epi==1: += bias[col] + sinusoidal PE(pos=row%2048, col)   (embedding epilogue)
// Block: 128x128 tile, 256 threads = 8 waves (4x2), wave = 32x64 (8 accums).
// Operand streaming: double-buffered LDS fed by global_load_async_to_lds_b128.

#define PE_COEF (-0.0089944730f)   // -ln(10000)/1024

__global__ __launch_bounds__(256) void gemm_bf16_wmma(
    const unsigned short* __restrict__ A,
    const unsigned short* __restrict__ Bt,
    unsigned short* __restrict__ C,
    int M, int N, int K,
    long long BzStride, long long CzStride,
    int epi, const float* __restrict__ bias)
{
  __shared__ alignas(16) unsigned short As[2][128 * TSTR];
  __shared__ alignas(16) unsigned short Bs[2][128 * TSTR];

  Bt += (long long)blockIdx.z * BzStride;
  C  += (long long)blockIdx.z * CzStride;

  int tid = threadIdx.x, lane = tid & 31, w = tid >> 5;
  int wm = w >> 1, wn = w & 1;                 // wave grid 4(M) x 2(N)
  int row0 = blockIdx.y * 128, col0 = blockIdx.x * 128;

  v8f acc[2][4];
  for (int mt = 0; mt < 2; ++mt)
    for (int nt = 0; nt < 4; ++nt)
      for (int j = 0; j < 8; ++j) acc[mt][nt][j] = 0.f;

  int r = tid >> 1, cs = tid & 1;              // 128 rows x two 16-elem segments
  const unsigned short* gA = A  + (size_t)(row0 + r) * K + cs * 16;
  const unsigned short* gB = Bt + (size_t)(col0 + r) * K + cs * 16;
  int lofs = r * TSTR + cs * 16;

  auto issue = [&](int kelem, int buf) {
    async_ld_b128(gA + kelem,     &As[buf][lofs]);
    async_ld_b128(gA + kelem + 8, &As[buf][lofs + 8]);
    async_ld_b128(gB + kelem,     &Bs[buf][lofs]);
    async_ld_b128(gB + kelem + 8, &Bs[buf][lofs + 8]);
  };

  int nk = K >> 5;
  issue(0, 0);
  for (int i = 0; i < nk; ++i) {
    int cur = i & 1;
    if (i + 1 < nk) {
      issue((i + 1) << 5, cur ^ 1);
      __builtin_amdgcn_s_wait_asynccnt(4);   // tile i complete (in-order retire)
    } else {
      __builtin_amdgcn_s_wait_asynccnt(0);
    }
    __syncthreads();

    v16bf af[2];
    af[0] = load_frag_a(As[cur], wm * 32,      lane);
    af[1] = load_frag_a(As[cur], wm * 32 + 16, lane);
    v16bf bf[4];
    for (int nt = 0; nt < 4; ++nt) bf[nt] = load_frag_b(Bs[cur], wn * 64 + nt * 16, lane);

    for (int mt = 0; mt < 2; ++mt)
      for (int nt = 0; nt < 4; ++nt)
        acc[mt][nt] = __builtin_amdgcn_wmma_f32_16x16x32_bf16(
            false, af[mt], false, bf[nt], (short)0, acc[mt][nt], false, false);
    __syncthreads();                          // protect buffer reused next issue
  }

  // epilogue: C/D layout -> lane col = lane&15, row = j + 8*(lane>>4)
  int halfc = lane >> 4;
  for (int mt = 0; mt < 2; ++mt)
    for (int nt = 0; nt < 4; ++nt)
      for (int j = 0; j < 8; ++j) {
        int gr = row0 + wm * 32 + mt * 16 + halfc * 8 + j;
        int gc = col0 + wn * 64 + nt * 16 + (lane & 15);
        float v = acc[mt][nt][j];
        if (epi == 1) {
          int pos = gr & 2047;
          float freq = __expf((float)(gc & ~1) * PE_COEF);
          float ang  = (float)pos * freq;
          v += bias[gc] + ((gc & 1) ? __cosf(ang) : __sinf(ang));
        }
        C[(size_t)gr * N + gc] = f2bf(v);
      }
}

// ------------------------------ V projection -------------------------------
// V[(b*16+h)*2048 + n] = sum_d h[b,n,d] * Wv[h,d,0]; one block per row r=b*2048+n.
__global__ __launch_bounds__(256) void v_proj(const unsigned short* __restrict__ Hbf,
                                              const float* __restrict__ Wv,
                                              float* __restrict__ V)
{
  __shared__ float hrow[1024];
  __shared__ float part[256];
  int r = blockIdx.x, tid = threadIdx.x;
  for (int i = 0; i < 4; ++i)
    hrow[tid * 4 + i] = bf2f(Hbf[(size_t)r * 1024 + tid * 4 + i]);
  __syncthreads();
  int head = tid >> 4, seg = tid & 15;
  const float* wv = Wv + head * 1024 + seg * 64;
  float p = 0.f;
  for (int d = 0; d < 64; ++d) p += hrow[seg * 64 + d] * wv[d];
  part[tid] = p;
  __syncthreads();
  if (tid < 16) {
    float s = 0.f;
    for (int k = 0; k < 16; ++k) s += part[tid * 16 + k];
    int b = r >> 11, n = r & 2047;
    V[((size_t)(b * 16 + tid)) * 2048 + n] = s;
  }
}

// ------------------------------- attention ---------------------------------
// grid (16 q-blocks, 32 b*h). Flash-style: 128x128 score tiles (WMMA bf16),
// async-LDS double-buffered Q/K streaming, online softmax with scalar V.
__global__ __launch_bounds__(256) void attn_kernel(
    const unsigned short* __restrict__ Qb,   // [head][4096][1024] bf16
    const unsigned short* __restrict__ Kb,   // [head][4096][1024] bf16
    const float* __restrict__ Vw,            // [(b*16+h)][2048]
    float* __restrict__ ctx)                 // [4096][16]
{
  __shared__ alignas(16) unsigned short Qs[2][128 * TSTR];
  __shared__ alignas(16) unsigned short Ks[2][128 * TSTR];
  __shared__ alignas(16) float Sc[128 * 132];   // padded: conflict-free writes
  __shared__ alignas(16) float Vs[128];

  int tid = threadIdx.x, lane = tid & 31, wq = tid >> 5;
  int qb = blockIdx.x, bh = blockIdx.y;
  int b = bh >> 4, h = bh & 15;
  const unsigned short* Qh  = Qb + (size_t)h * 4096 * 1024;
  const unsigned short* Kh  = Kb + (size_t)h * 4096 * 1024;
  const float*          Vbh = Vw + (size_t)bh * 2048;
  int qrow0 = b * 2048 + qb * 128;
  int krowB = b * 2048;

  float m = -3.0e38f, l = 0.f, acc = 0.f;
  int r = tid >> 1, cs = tid & 1;
  int lofs = r * TSTR + cs * 16;
  const unsigned short* gQ = Qh + (size_t)(qrow0 + r) * 1024 + cs * 16;

  for (int kb = 0; kb < 16; ++kb) {
    const unsigned short* gK = Kh + (size_t)(krowB + kb * 128 + r) * 1024 + cs * 16;

    v8f sacc[8];
    for (int t = 0; t < 8; ++t)
      for (int j = 0; j < 8; ++j) sacc[t][j] = 0.f;

    auto issue = [&](int kelem, int buf) {
      async_ld_b128(gQ + kelem,     &Qs[buf][lofs]);
      async_ld_b128(gQ + kelem + 8, &Qs[buf][lofs + 8]);
      async_ld_b128(gK + kelem,     &Ks[buf][lofs]);
      async_ld_b128(gK + kelem + 8, &Ks[buf][lofs + 8]);
    };

    issue(0, 0);
    for (int et = 0; et < 32; ++et) {        // contraction over e (1024)
      int cur = et & 1;
      if (et + 1 < 32) {
        issue((et + 1) << 5, cur ^ 1);
        __builtin_amdgcn_s_wait_asynccnt(4);
      } else {
        __builtin_amdgcn_s_wait_asynccnt(0);
      }
      __syncthreads();

      v16bf af = load_frag_a(Qs[cur], wq * 16, lane);  // wave: 16 q-rows x 128 keys
      for (int t = 0; t < 8; ++t) {
        v16bf bf = load_frag_b(Ks[cur], t * 16, lane);
        sacc[t] = __builtin_amdgcn_wmma_f32_16x16x32_bf16(
            false, af, false, bf, (short)0, sacc[t], false, false);
      }
      __syncthreads();
    }

    // scale by 1/sqrt(1024), mask global key 0, write score tile to LDS
    int halfc = lane >> 4;
    for (int t = 0; t < 8; ++t) {
      int ccol = t * 16 + (lane & 15);
      int gk0  = kb * 128 + ccol;
      for (int j = 0; j < 8; ++j) {
        int rrow = wq * 16 + halfc * 8 + j;
        float v = sacc[t][j] * 0.03125f;
        if (gk0 == 0) v = 0.f;
        Sc[rrow * 132 + ccol] = v;
      }
    }
    if (tid < 128) Vs[tid] = Vbh[kb * 128 + tid];
    __syncthreads();

    if (tid < 128) {                          // online softmax, one thread per q row
      const f32x4* srow = (const f32x4*)(Sc + tid * 132);
      const f32x4* vv   = (const f32x4*)Vs;
      float tmax = -3.0e38f;
      for (int k4 = 0; k4 < 32; ++k4) {
        f32x4 s = srow[k4];
        tmax = fmaxf(tmax, fmaxf(fmaxf(s[0], s[1]), fmaxf(s[2], s[3])));
      }
      float nm = fmaxf(m, tmax);
      float scale = __expf(m - nm);
      l *= scale; acc *= scale;
      for (int k4 = 0; k4 < 32; ++k4) {
        f32x4 s = srow[k4];
        f32x4 v = vv[k4];
        float e0 = __expf(s[0] - nm), e1 = __expf(s[1] - nm);
        float e2 = __expf(s[2] - nm), e3 = __expf(s[3] - nm);
        l   += (e0 + e1) + (e2 + e3);
        acc += (e0 * v[0] + e1 * v[1]) + (e2 * v[2] + e3 * v[3]);
      }
      m = nm;
    }
    __syncthreads();
  }

  if (tid < 128) ctx[(size_t)(qrow0 + tid) * 16 + h] = acc / l;
}

// --------------------------- fused tail GEMMs ------------------------------

__global__ void wo_dec(const float* __restrict__ Wo, const float* __restrict__ dec_w,
                       float* __restrict__ M16) {
  int gid = blockIdx.x * 256 + threadIdx.x;           // 16*1024
  int j = gid >> 10, c = gid & 1023;
  float s = 0.f;
  for (int d = 0; d < 1024; ++d) s += Wo[j * 1024 + d] * dec_w[(size_t)d * 1024 + c];
  M16[gid] = s;
}

__global__ void final_k(const float* __restrict__ ctx, const float* __restrict__ M16,
                        const float* __restrict__ dec_b, float* __restrict__ out) {
  int gid = blockIdx.x * 256 + threadIdx.x;           // 4096*1024
  int rI = gid >> 10, c = gid & 1023;
  float s = dec_b[c];
  for (int j = 0; j < 16; ++j) s += ctx[rI * 16 + j] * M16[j * 1024 + c];
  out[gid] = s;
}

// ------------------------------- launcher ----------------------------------

extern "C" void kernel_launch(void* const* d_in, const int* in_sizes, int n_in,
                              void* d_out, int out_size, void* d_ws, size_t ws_size,
                              hipStream_t stream) {
  const float* X     = (const float*)d_in[0];
  const float* emb_w = (const float*)d_in[1];
  const float* emb_b = (const float*)d_in[2];
  const float* Wq    = (const float*)d_in[3];
  const float* Wk    = (const float*)d_in[4];
  const float* Wv    = (const float*)d_in[5];
  const float* Wo    = (const float*)d_in[6];
  const float* dec_w = (const float*)d_in[7];
  const float* dec_b = (const float*)d_in[8];
  float* out = (float*)d_out;

  char* ws = (char*)d_ws;
  size_t o = 0;
  auto alloc = [&](size_t bytes) -> void* {
    void* p = ws + o;
    o += (bytes + 255) & ~(size_t)255;
    return p;
  };
  unsigned short* Xbf = (unsigned short*)alloc(4096ull * 512 * 2);
  unsigned short* EWt = (unsigned short*)alloc(1024ull * 512 * 2);          // emb_w^T bf16
  unsigned short* Hbf = (unsigned short*)alloc(4096ull * 1024 * 2);
  unsigned short* Wqt = (unsigned short*)alloc(16ull * 1024 * 1024 * 2);    // per-head W^T
  unsigned short* Wkt = (unsigned short*)alloc(16ull * 1024 * 1024 * 2);
  unsigned short* Qb  = (unsigned short*)alloc(16ull * 4096 * 1024 * 2);
  unsigned short* Kb  = (unsigned short*)alloc(16ull * 4096 * 1024 * 2);
  float* Vw  = (float*)alloc(32ull * 2048 * 4);
  float* ctx = (float*)alloc(4096ull * 16 * 4);
  float* M16 = (float*)alloc(16ull * 1024 * 4);

  // 1) conversions / transposes
  cvt_f32_bf16<<<(4096 * 512) / 256, 256, 0, stream>>>(X, Xbf, 4096 * 512);
  transpose_cvt<<<dim3(32, 16, 1), dim3(32, 8), 0, stream>>>(emb_w, EWt, 512, 1024, 0, 0);
  transpose_cvt<<<dim3(32, 32, 16), dim3(32, 8), 0, stream>>>(Wq, Wqt, 1024, 1024, 1048576, 1048576);
  transpose_cvt<<<dim3(32, 32, 16), dim3(32, 8), 0, stream>>>(Wk, Wkt, 1024, 1024, 1048576, 1048576);

  // 2) embedding GEMM + bias + PE -> h (bf16)
  gemm_bf16_wmma<<<dim3(8, 32, 1), 256, 0, stream>>>(Xbf, EWt, Hbf, 4096, 1024, 512,
                                                     0, 0, 1, emb_b);
  // 3) V projection
  v_proj<<<4096, 256, 0, stream>>>(Hbf, Wv, Vw);

  // 4) Q / K projections (16-head batched)
  gemm_bf16_wmma<<<dim3(8, 32, 16), 256, 0, stream>>>(Hbf, Wqt, Qb, 4096, 1024, 1024,
                                                      1048576, 4194304, 0, nullptr);
  gemm_bf16_wmma<<<dim3(8, 32, 16), 256, 0, stream>>>(Hbf, Wkt, Kb, 4096, 1024, 1024,
                                                      1048576, 4194304, 0, nullptr);

  // 5) fused tail weight: M16 = Wo @ dec_w
  wo_dec<<<64, 256, 0, stream>>>(Wo, dec_w, M16);

  // 6) attention (flash-style, scalar V)
  attn_kernel<<<dim3(16, 32), 256, 0, stream>>>(Qb, Kb, Vw, ctx);

  // 7) out = ctx @ M16 + dec_b
  final_k<<<(4096 * 1024) / 256, 256, 0, stream>>>(ctx, M16, dec_b, out);
}
